// BatchProgramClassifier_74414603371109
// MI455X (gfx1250) — matrix operand order repaired and checked by reading, not compiled, not run on hardware
//
#include <hip/hip_runtime.h>
#include <hip/hip_bf16.h>

typedef __attribute__((ext_vector_type(16))) __bf16 v16bf;
typedef __attribute__((ext_vector_type(8)))  __bf16 v8bf;
typedef __attribute__((ext_vector_type(8)))  float  v8f;

#define B_   128
#define L_   128
#define N_   16
#define E_   128
#define D_   128
#define H_   100
#define G3   300          // 3H
#define GP   304          // 3H padded to 19 WMMA tiles
#define C_   104
#define T_   (B_ * L_)    // 16384 statements

// ---- helpers ---------------------------------------------------------------

__device__ __forceinline__ __bf16 f2bf(float f) {
    union { float f; unsigned u; } v; v.f = f;
    unsigned r = (v.u + 0x7FFFu + ((v.u >> 16) & 1u)) >> 16;   // RNE truncate
    union { unsigned short s; __bf16 b; } o; o.s = (unsigned short)r;
    return o.b;
}

__device__ __forceinline__ v8f wmma_bf16(v16bf a, v16bf b, v8f c) {
    // (neg_a, A, neg_b, B, c_mod, C, reuse_a, reuse_b)
    return __builtin_amdgcn_wmma_f32_16x16x32_bf16(false, a, false, b,
                                                   (short)0, c, false, false);
}

// A fragment (16x32, bf16) built from a contiguous f32 row.
// ISA layout: lanes 0-15 hold K={0..7,16..23}, lanes 16-31 K={8..15,24..31}.
__device__ __forceinline__ v16bf a_frag_f32(const float* row, int kc, int lane) {
    const int base = kc * 32 + ((lane >> 4) << 3);
    v16bf a;
#pragma unroll
    for (int h = 0; h < 2; ++h) {
        const float4* p = (const float4*)(row + base + h * 16);
        float4 v0 = p[0], v1 = p[1];
        a[h*8+0] = f2bf(v0.x); a[h*8+1] = f2bf(v0.y);
        a[h*8+2] = f2bf(v0.z); a[h*8+3] = f2bf(v0.w);
        a[h*8+4] = f2bf(v1.x); a[h*8+5] = f2bf(v1.y);
        a[h*8+6] = f2bf(v1.z); a[h*8+7] = f2bf(v1.w);
    }
    return a;
}

// A fragment from a bf16 row (LDS-resident hidden state).
__device__ __forceinline__ v16bf a_frag_bf16(const __bf16* row, int kc, int lane) {
    const int base = kc * 32 + ((lane >> 4) << 3);
    v8bf lo = *(const v8bf*)(row + base);
    v8bf hi = *(const v8bf*)(row + base + 16);
    v16bf a;
#pragma unroll
    for (int e = 0; e < 8; ++e) { a[e] = lo[e]; a[8 + e] = hi[e]; }
    return a;
}

// B fragment (32x16, bf16): lane holds column n; lanes 0-15 K={0..7,16..23},
// lanes 16-31 K={8..15,24..31} (two 16B loads).
__device__ __forceinline__ v16bf b_frag_bf16(const __bf16* wrow, int kc, int lane) {
    const int kb = kc * 32 + ((lane >> 4) << 3);
    v8bf lo = *(const v8bf*)(wrow + kb);
    v8bf hi = *(const v8bf*)(wrow + kb + 16);
    v16bf b;
#pragma unroll
    for (int e = 0; e < 8; ++e) { b[e] = lo[e]; b[8 + e] = hi[e]; }
    return b;
}

// ---- kernel 0: weight pre-conversion / padding to bf16 ---------------------

__global__ __launch_bounds__(256) void k_prep(
    const float* __restrict__ Wc_w,
    const float* __restrict__ Wih_f, const float* __restrict__ Wih_b,
    const float* __restrict__ Whh_f, const float* __restrict__ Whh_b,
    __bf16* __restrict__ wc_bf, __bf16* __restrict__ wih_pad,
    __bf16* __restrict__ whh_pad)
{
    const int i = blockIdx.x * 256 + threadIdx.x;
    if (i < D_ * E_) wc_bf[i] = f2bf(Wc_w[i]);
    if (i < 2 * GP * D_) {
        const int d = i / (GP * D_), rem = i % (GP * D_);
        const int r = rem / D_, k = rem % D_;
        const float* Wih = d ? Wih_b : Wih_f;
        const float* Whh = d ? Whh_b : Whh_f;
        wih_pad[i] = (r < G3) ? f2bf(Wih[r * D_ + k]) : f2bf(0.0f);
        whh_pad[i] = (r < G3 && k < H_) ? f2bf(Whh[r * H_ + k]) : f2bf(0.0f);
    }
}

// ---- kernel 1: fused gather + projection GEMM + tree aggregation + pool ----
// One wave per statement tree (16 nodes == one WMMA M-tile).
// kc-outer / nt-inner: 8 live accumulators, batched B loads per kc.

__global__ __launch_bounds__(128) void k_tree_encode(
    const int* __restrict__ tokens, const int* __restrict__ parents,
    const float* __restrict__ emb, const __bf16* __restrict__ wc_bf,
    const float* __restrict__ Wc_b, float* __restrict__ enc)
{
    __shared__ float sT[4][N_][D_];                 // 32 KB: per-wave tree tile
    const int tid = threadIdx.x, lane = tid & 31, wave = tid >> 5;
    const int t = blockIdx.x * 4 + wave;            // tree index (b*L + l)
    const int mr = lane & 15, mh = (lane >> 4) << 3;

    const int tok = tokens[t * N_ + mr];
    const float* arow = emb + (size_t)tok * E_;
    v16bf A[4];
#pragma unroll
    for (int kc = 0; kc < 4; ++kc) A[kc] = a_frag_f32(arow, kc, lane);

    v8f acc[8];
#pragma unroll
    for (int nt = 0; nt < 8; ++nt) {
        const float bias = Wc_b[nt * 16 + mr];
#pragma unroll
        for (int i = 0; i < 8; ++i) acc[nt][i] = bias;
    }
#pragma unroll
    for (int kc = 0; kc < 4; ++kc) {
        v16bf Bk[8];
#pragma unroll
        for (int nt = 0; nt < 8; ++nt)
            Bk[nt] = b_frag_bf16(wc_bf + (size_t)(nt * 16 + mr) * E_, kc, lane);
#pragma unroll
        for (int nt = 0; nt < 8; ++nt)
            acc[nt] = wmma_bf16(A[kc], Bk[nt], acc[nt]);
    }
#pragma unroll
    for (int nt = 0; nt < 8; ++nt) {
#pragma unroll
        for (int r = 0; r < 8; ++r) sT[wave][mh + r][nt * 16 + mr] = acc[nt][r];
    }

    // bottom-up aggregation: h[parent] += h[child], j = 15..1 (sequential),
    // each lane owns 4 contiguous columns.
    const int c0 = lane * 4;
    for (int j = N_ - 1; j >= 1; --j) {
        const int p = parents[t * N_ + j];
        float4 v = *(const float4*)&sT[wave][j][c0];
        float4 u = *(float4*)&sT[wave][p][c0];
        u.x += v.x; u.y += v.y; u.z += v.z; u.w += v.w;
        *(float4*)&sT[wave][p][c0] = u;
    }

    // enc = relu(max over nodes)
    float4 mx = make_float4(0.f, 0.f, 0.f, 0.f);
#pragma unroll
    for (int m = 0; m < N_; ++m) {
        float4 v = *(const float4*)&sT[wave][m][c0];
        mx.x = fmaxf(mx.x, v.x); mx.y = fmaxf(mx.y, v.y);
        mx.z = fmaxf(mx.z, v.z); mx.w = fmaxf(mx.w, v.w);
    }
    *(float4*)&enc[(size_t)t * D_ + c0] = mx;
}

// ---- kernel 2: GRU input projection xW = enc @ Wih^T + bih (both dirs) -----
// B fragments double-buffered across the 19 N-tiles.

__global__ __launch_bounds__(256) void k_xw(
    const float* __restrict__ enc, const __bf16* __restrict__ wih_pad,
    const float* __restrict__ bih_f, const float* __restrict__ bih_b,
    float* __restrict__ xw_f, float* __restrict__ xw_b)
{
    const int tid = threadIdx.x, lane = tid & 31, wave = tid >> 5;
    const int ww = blockIdx.x * 8 + wave;           // 2048 wave tasks
    const int dir = ww & 1, rt = ww >> 1;           // row tile 0..1023
    const float* bih = dir ? bih_b : bih_f;
    const __bf16* wih = wih_pad + (size_t)dir * GP * D_;
    float* xw = dir ? xw_b : xw_f;
    const int mr = lane & 15, mh = (lane >> 4) << 3;

    const float* arow = enc + (size_t)(rt * 16 + mr) * D_;
    v16bf A[4];
#pragma unroll
    for (int kc = 0; kc < 4; ++kc) A[kc] = a_frag_f32(arow, kc, lane);

    v16bf bnext[4];
#pragma unroll
    for (int kc = 0; kc < 4; ++kc)
        bnext[kc] = b_frag_bf16(wih + (size_t)mr * D_, kc, lane);

    for (int nt = 0; nt < 19; ++nt) {
        v16bf bcur[4];
#pragma unroll
        for (int kc = 0; kc < 4; ++kc) bcur[kc] = bnext[kc];
        if (nt < 18) {
            const __bf16* wrow = wih + (size_t)((nt + 1) * 16 + mr) * D_;
#pragma unroll
            for (int kc = 0; kc < 4; ++kc) bnext[kc] = b_frag_bf16(wrow, kc, lane);
        }
        const int n = nt * 16 + mr;
        const float bias = (n < G3) ? bih[n] : 0.f;
        v8f c;
#pragma unroll
        for (int i = 0; i < 8; ++i) c[i] = bias;
#pragma unroll
        for (int kc = 0; kc < 4; ++kc)
            c = wmma_bf16(A[kc], bcur[kc], c);
        if (n < G3) {
#pragma unroll
            for (int r = 0; r < 8; ++r)
                xw[(size_t)(rt * 16 + mh + r) * GP + n] = c[r];
        }
    }
}

// ---- kernel 3: bidirectional GRU recurrence + time max-pool ----------------
// blockIdx: dir = bi>>3, batch tile = bi&7 (16 batch rows). 8 waves.
// Whh B-fragments are loop-invariant -> held in registers across all steps.

__global__ __launch_bounds__(256) void k_gru(
    const float* __restrict__ xw_f, const float* __restrict__ xw_b,
    const __bf16* __restrict__ whh_pad,
    const float* __restrict__ bhh_f, const float* __restrict__ bhh_b,
    float* __restrict__ pooled)
{
    __shared__ __bf16 sH[N_ * D_];      // hprev bf16, [16][128] K-padded, 4 KB
    __shared__ float  sG[N_ * GP];      // hW gate buffer [16][304], 19 KB
    const int dir = blockIdx.x >> 3;
    const int b0  = (blockIdx.x & 7) * 16;
    const int tid = threadIdx.x, lane = tid & 31, wave = tid >> 5;
    const int mr = lane & 15, mh = (lane >> 4) << 3;
    const float* xw  = dir ? xw_b : xw_f;
    const float* bhh = dir ? bhh_b : bhh_f;
    const __bf16* whh = whh_pad + (size_t)dir * GP * D_;

    // register-resident Whh fragments: wave handles n-tiles {wave, wave+8, wave+16}
    v16bf Bf[3][4];
    float bias3[3];
#pragma unroll
    for (int i = 0; i < 3; ++i) {
        const int nt = wave + i * 8;
        if (nt < 19) {
            const int n = nt * 16 + mr;
            bias3[i] = (n < G3) ? bhh[n] : 0.f;
            const __bf16* wrow = whh + (size_t)n * D_;
#pragma unroll
            for (int kc = 0; kc < 4; ++kc) Bf[i][kc] = b_frag_bf16(wrow, kc, lane);
        }
    }

    for (int i = tid; i < N_ * D_; i += 256) sH[i] = f2bf(0.0f);
    float hcur[7], rmax[7];
#pragma unroll
    for (int it = 0; it < 7; ++it) { hcur[it] = 0.f; rmax[it] = -3.4e38f; }
    __syncthreads();

    for (int step = 0; step < L_; ++step) {
        const int l = dir ? (L_ - 1 - step) : step;

        // warm L2/L0 for this step's gate inputs while the GEMM runs
#pragma unroll
        for (int it = 0; it < 7; ++it) {
            const int e = tid + it * 256;
            if (e < N_ * H_) {
                const int m = e / H_, u = e % H_;
                const float* x = xw + ((size_t)(b0 + m) * L_ + l) * GP;
                __builtin_prefetch(x + u, 0, 0);
                __builtin_prefetch(x + H_ + u, 0, 0);
                __builtin_prefetch(x + 2 * H_ + u, 0, 0);
            }
        }

        // hW = hprev @ Whh^T + bhh  (K padded 100->128, N padded 300->304)
        v16bf A[4];
#pragma unroll
        for (int kc = 0; kc < 4; ++kc) A[kc] = a_frag_bf16(&sH[mr * D_], kc, lane);
#pragma unroll
        for (int i = 0; i < 3; ++i) {
            const int nt = wave + i * 8;
            if (nt < 19) {
                const int n = nt * 16 + mr;
                v8f c;
#pragma unroll
                for (int j = 0; j < 8; ++j) c[j] = bias3[i];
#pragma unroll
                for (int kc = 0; kc < 4; ++kc)
                    c = wmma_bf16(A[kc], Bf[i][kc], c);
#pragma unroll
                for (int r = 0; r < 8; ++r) sG[(mh + r) * GP + n] = c[r];
            }
        }
        __syncthreads();

        // gates: 16 rows x 100 units = 1600 elems over 256 threads
#pragma unroll
        for (int it = 0; it < 7; ++it) {
            const int e = tid + it * 256;
            if (e < N_ * H_) {
                const int m = e / H_, u = e % H_;
                const float* x = xw + ((size_t)(b0 + m) * L_ + l) * GP;
                const float xr = x[u], xz = x[H_ + u], xn = x[2 * H_ + u];
                const float hr = sG[m * GP + u];
                const float hz = sG[m * GP + H_ + u];
                const float hn = sG[m * GP + 2 * H_ + u];
                const float r  = 1.f / (1.f + __expf(-(xr + hr)));
                const float z  = 1.f / (1.f + __expf(-(xz + hz)));
                const float nn = tanhf(xn + r * hn);
                const float h  = (1.f - z) * nn + z * hcur[it];
                hcur[it] = h;
                rmax[it] = fmaxf(rmax[it], h);
                sH[m * D_ + u] = f2bf(h);
            }
        }
        __syncthreads();
    }

#pragma unroll
    for (int it = 0; it < 7; ++it) {
        const int e = tid + it * 256;
        if (e < N_ * H_) {
            const int m = e / H_, u = e % H_;
            pooled[(size_t)(b0 + m) * (2 * H_) + dir * H_ + u] = rmax[it];
        }
    }
}

// ---- kernel 4: logits = pooled @ lbl_w^T + lbl_b (tiny, fp32 exact) --------

__global__ __launch_bounds__(256) void k_logits(
    const float* __restrict__ pooled, const float* __restrict__ lbl_w,
    const float* __restrict__ lbl_b, float* __restrict__ out)
{
    const int i = blockIdx.x * 256 + threadIdx.x;
    if (i >= B_ * C_) return;
    const int b = i / C_, c = i % C_;
    const float* p = pooled + (size_t)b * (2 * H_);
    const float* w = lbl_w + (size_t)c * (2 * H_);
    float s = lbl_b[c];
    for (int k = 0; k < 2 * H_; ++k) s += p[k] * w[k];
    out[i] = s;
}

// ---- host glue -------------------------------------------------------------

extern "C" void kernel_launch(void* const* d_in, const int* in_sizes, int n_in,
                              void* d_out, int out_size, void* d_ws, size_t ws_size,
                              hipStream_t stream)
{
    (void)in_sizes; (void)n_in; (void)out_size; (void)ws_size;
    const int*   tokens  = (const int*)d_in[0];
    const int*   parents = (const int*)d_in[1];
    const float* emb     = (const float*)d_in[2];
    const float* Wc_w    = (const float*)d_in[3];
    const float* Wc_b    = (const float*)d_in[4];
    const float* Wih_f   = (const float*)d_in[5];
    const float* Whh_f   = (const float*)d_in[6];
    const float* bih_f   = (const float*)d_in[7];
    const float* bhh_f   = (const float*)d_in[8];
    const float* Wih_b   = (const float*)d_in[9];
    const float* Whh_b   = (const float*)d_in[10];
    const float* bih_b   = (const float*)d_in[11];
    const float* bhh_b   = (const float*)d_in[12];
    const float* lbl_w   = (const float*)d_in[13];
    const float* lbl_b   = (const float*)d_in[14];

    char* ws = (char*)d_ws;
    const size_t ENC = 0;                                  // 16384*128 f32
    const size_t XWF = ENC + (size_t)T_ * D_ * 4;          // 16384*304 f32
    const size_t XWB = XWF + (size_t)T_ * GP * 4;
    const size_t WCB = XWB + (size_t)T_ * GP * 4;          // 128*128 bf16
    const size_t WIH = WCB + (size_t)D_ * E_ * 2;          // 2*304*128 bf16
    const size_t WHH = WIH + (size_t)2 * GP * D_ * 2;
    const size_t PLD = WHH + (size_t)2 * GP * D_ * 2;      // 128*200 f32

    float*  enc     = (float*) (ws + ENC);
    float*  xw_f    = (float*) (ws + XWF);
    float*  xw_b    = (float*) (ws + XWB);
    __bf16* wc_bf   = (__bf16*)(ws + WCB);
    __bf16* wih_pad = (__bf16*)(ws + WIH);
    __bf16* whh_pad = (__bf16*)(ws + WHH);
    float*  pooled  = (float*) (ws + PLD);

    k_prep<<<(2 * GP * D_ + 255) / 256, 256, 0, stream>>>(
        Wc_w, Wih_f, Wih_b, Whh_f, Whh_b, wc_bf, wih_pad, whh_pad);
    k_tree_encode<<<T_ / 4, 128, 0, stream>>>(tokens, parents, emb, wc_bf, Wc_b, enc);
    k_xw<<<(T_ / 16) * 2 / 8, 256, 0, stream>>>(enc, wih_pad, bih_f, bih_b, xw_f, xw_b);
    k_gru<<<16, 256, 0, stream>>>(xw_f, xw_b, whh_pad, bhh_f, bhh_b, pooled);
    k_logits<<<(B_ * C_ + 255) / 256, 256, 0, stream>>>(pooled, lbl_w, lbl_b, (float*)d_out);
}
